// LinearAttention_39951785787407
// MI455X (gfx1250) — matrix operand order repaired and checked by reading, not compiled, not run on hardware
//
#include <hip/hip_runtime.h>

// ---------------------------------------------------------------------------
// LinearAttention on MI455X (gfx1250): f16 WMMA pipeline, fp32 accumulate.
// All GEMM operands are laid out K-contiguous so every WMMA fragment is two
// 16-byte loads (no LDS staging / no per-element shuffles).
// ---------------------------------------------------------------------------
typedef __attribute__((ext_vector_type(16))) _Float16 v16h;
typedef __attribute__((ext_vector_type(8)))  float    v8f;
typedef __attribute__((ext_vector_type(4)))  int      vi4;

union Frag {
    v16h     v;
    uint4    q[2];
    _Float16 h[16];
};
static_assert(sizeof(Frag) == 32, "frag size");

#define WMMA_F16(A, B, C) \
    __builtin_amdgcn_wmma_f32_16x16x32_f16(false, (A), false, (B), (short)0, (C), false, false)

constexpr int BATCH   = 16;
constexpr int C_IN    = 256;
constexpr int NPIX    = 4096;      // 64*64
constexpr int HEADS   = 4;
constexpr int DHEAD   = 32;
constexpr int HID     = 128;       // HEADS*DHEAD
constexpr int OC_QKV  = 384;       // 3*HID

// workspace layout (bytes)
constexpr size_t OFF_XT    = 0;                                    // f16 xT[b][n][c]
constexpr size_t OFF_WQKV  = OFF_XT   + (size_t)BATCH*C_IN*NPIX*2;
constexpr size_t OFF_WOUT  = OFF_WQKV + (size_t)OC_QKV*C_IN*2;
constexpr size_t OFF_QKV   = OFF_WOUT + (size_t)C_IN*HID*2;        // f16 qkv[b][o][n]
constexpr size_t OFF_QT    = OFF_QKV  + (size_t)BATCH*OC_QKV*NPIX*2; // f16 qT[b][h][n][d]
constexpr size_t OFF_CTX   = OFF_QT   + (size_t)BATCH*HID*NPIX*2;  // f16 ctxT[b][h][e][d]
constexpr size_t OFF_ATT   = OFF_CTX  + (size_t)BATCH*HEADS*DHEAD*DHEAD*2; // f16 attT[b][n][ch]
constexpr size_t OFF_STATS = OFF_ATT  + (size_t)BATCH*NPIX*HID*2;  // f32 sums

// ---------------------------------------------------------------------------
// CDNA5 async global->LDS copy (ASYNCcnt path), with safe fallback.
// ---------------------------------------------------------------------------
#if defined(__HIP_DEVICE_COMPILE__) && \
    __has_builtin(__builtin_amdgcn_global_load_async_to_lds_b128) && \
    __has_builtin(__builtin_amdgcn_s_wait_asynccnt)
#define HAS_ASYNC_LDS 1
#else
#define HAS_ASYNC_LDS 0
#endif

__device__ __forceinline__ void async_copy16(const void* g, void* l) {
#if HAS_ASYNC_LDS
    __builtin_amdgcn_global_load_async_to_lds_b128(
        (__attribute__((address_space(1))) vi4*)g,
        (__attribute__((address_space(3))) vi4*)l, 0, 0);
#else
    *(uint4*)l = *(const uint4*)g;
#endif
}
__device__ __forceinline__ void async_join() {
#if HAS_ASYNC_LDS
    __builtin_amdgcn_s_wait_asynccnt(0);
#endif
}

// ---------------------------------------------------------------------------
__global__ __launch_bounds__(64) void zero_stats_kernel(float* stats) {
    if (threadIdx.x < 2 * BATCH) stats[threadIdx.x] = 0.0f;
}

// x[b][c][n] f32  ->  xT[b][n][c] f16   (LDS-tiled 32x32 transpose, async fill)
__global__ __launch_bounds__(256) void pack_x_kernel(const float* __restrict__ x,
                                                     _Float16* __restrict__ xT) {
    __shared__ float tile[32][36];                    // pad keeps 16B alignment
    const int b = blockIdx.z, c0 = blockIdx.y * 32, n0 = blockIdx.x * 32;
    const int t = threadIdx.x, row = t >> 3, q = t & 7;
    async_copy16(x + ((size_t)b * C_IN + c0 + row) * NPIX + n0 + q * 4,
                 &tile[row][q * 4]);
    async_join();
    __syncthreads();
    union { _Float16 h[4]; uint2 u; } p;
    #pragma unroll
    for (int j = 0; j < 4; ++j) p.h[j] = (_Float16)tile[q * 4 + j][row];
    *(uint2*)&xT[((size_t)b * NPIX + n0 + row) * C_IN + c0 + q * 4] = p.u;
}

__global__ __launch_bounds__(256) void pack_w_kernel(const float* __restrict__ wq,
                                                     const float* __restrict__ wo,
                                                     _Float16* __restrict__ wqh,
                                                     _Float16* __restrict__ woh) {
    size_t i = ((size_t)blockIdx.x * 256 + threadIdx.x) * 4;
    const float* src; _Float16* dst; size_t j;
    constexpr size_t NQ = (size_t)OC_QKV * C_IN;   // 98304
    if (i < NQ) { src = wq; dst = wqh; j = i; }
    else        { src = wo; dst = woh; j = i - NQ; }
    float4 v = *(const float4*)(src + j);
    union { _Float16 h[4]; uint2 u; } p;
    p.h[0] = (_Float16)v.x; p.h[1] = (_Float16)v.y;
    p.h[2] = (_Float16)v.z; p.h[3] = (_Float16)v.w;
    *(uint2*)(dst + j) = p.u;
}

// ---------------------------------------------------------------------------
// qkv[b][o][n] = wq(384x256) @ x(256xN), computed as out[n][o] tiles.
// A = xT rows (K=c contig), B = wq rows (K=c contig).  No LDS.
__global__ __launch_bounds__(256) void qkv_gemm_kernel(const _Float16* __restrict__ xT,
                                                       const _Float16* __restrict__ wq,
                                                       _Float16* __restrict__ qkv) {
    const int tid = threadIdx.x, lane = tid & 31, wave = tid >> 5;
    const int wm = wave & 3, wn = wave >> 2;          // 4 waves along n, 2 along o
    const int b = blockIdx.z, n0 = blockIdx.x * 64, o0 = blockIdx.y * 128;
    const int m = lane & 15, hf = lane >> 4;
    const _Float16* arow = xT + ((size_t)b * NPIX + n0 + wm * 16 + m) * C_IN;

    v8f acc[4] = {};
    #pragma unroll
    for (int kk = 0; kk < C_IN; kk += 32) {
        Frag fa;
        fa.q[0] = *(const uint4*)(arow + kk + hf * 8);
        fa.q[1] = *(const uint4*)(arow + kk + 16 + hf * 8);
        #pragma unroll
        for (int nn = 0; nn < 4; ++nn) {
            const int o = o0 + wn * 64 + nn * 16 + m;
            Frag fb;
            fb.q[0] = *(const uint4*)(wq + (size_t)o * C_IN + kk + hf * 8);
            fb.q[1] = *(const uint4*)(wq + (size_t)o * C_IN + kk + 16 + hf * 8);
            acc[nn] = WMMA_F16(fa.v, fb.v, acc[nn]);
        }
    }
    const int nst = n0 + wm * 16 + hf * 8;            // 8 consecutive n per lane
    #pragma unroll
    for (int nn = 0; nn < 4; ++nn) {
        const int o = o0 + wn * 64 + nn * 16 + m;
        union { _Float16 h[8]; uint4 u; } pk;
        #pragma unroll
        for (int r = 0; r < 8; ++r) pk.h[r] = (_Float16)acc[nn][r];
        *(uint4*)(qkv + ((size_t)b * OC_QKV + o) * NPIX + nst) = pk.u;
    }
}

// ---------------------------------------------------------------------------
// q: softmax over d (strided read), *1/sqrt(32); writes transposed qT[b,h,n,d]
__global__ __launch_bounds__(256) void q_softmax_kernel(const _Float16* __restrict__ qkv,
                                                        _Float16* __restrict__ qT) {
    const int n = blockIdx.x * 256 + threadIdx.x;
    const int h = blockIdx.y, b = blockIdx.z;
    const _Float16* base = qkv + ((size_t)b * OC_QKV + h * DHEAD) * NPIX + n;
    float v[DHEAD], mx = -1e30f;
    #pragma unroll
    for (int d = 0; d < DHEAD; ++d) { v[d] = (float)base[(size_t)d * NPIX]; mx = fmaxf(mx, v[d]); }
    float s = 0.0f;
    #pragma unroll
    for (int d = 0; d < DHEAD; ++d) { v[d] = __expf(v[d] - mx); s += v[d]; }
    const float inv = 0.17677669529663687f / s;       // scale / sum
    union { _Float16 h[DHEAD]; uint4 u[4]; } p;
    #pragma unroll
    for (int d = 0; d < DHEAD; ++d) p.h[d] = (_Float16)(v[d] * inv);
    _Float16* dst = qT + (((size_t)b * HEADS + h) * NPIX + n) * DHEAD;
    #pragma unroll
    for (int j = 0; j < 4; ++j) *(uint4*)(dst + j * 8) = p.u[j];
}

// k: softmax over n (4096); one WG per (b, channel) row; in place.
__global__ __launch_bounds__(256) void k_softmax_kernel(_Float16* qkv) {
    __shared__ float red[256];
    const int row = blockIdx.x;                       // 0..2047
    const int b = row >> 7, o = HID + (row & 127);
    _Float16* base = qkv + ((size_t)b * OC_QKV + o) * NPIX;
    const int tid = threadIdx.x;
    float v[16], mx = -1e30f;
    #pragma unroll
    for (int i = 0; i < 16; ++i) { v[i] = (float)base[tid + i * 256]; mx = fmaxf(mx, v[i]); }
    red[tid] = mx; __syncthreads();
    for (int s = 128; s > 0; s >>= 1) { if (tid < s) red[tid] = fmaxf(red[tid], red[tid + s]); __syncthreads(); }
    mx = red[0]; __syncthreads();
    float sum = 0.0f;
    #pragma unroll
    for (int i = 0; i < 16; ++i) { v[i] = __expf(v[i] - mx); sum += v[i]; }
    red[tid] = sum; __syncthreads();
    for (int s = 128; s > 0; s >>= 1) { if (tid < s) red[tid] += red[tid + s]; __syncthreads(); }
    const float inv = 1.0f / red[0];
    #pragma unroll
    for (int i = 0; i < 16; ++i) base[tid + i * 256] = (_Float16)(v[i] * inv);
}

// ---------------------------------------------------------------------------
// ctxT[b,h,e,d] = sum_n k[d,n] * v[e,n].  K=4096 split over 8 waves.
__global__ __launch_bounds__(256) void ctx_gemm_kernel(const _Float16* __restrict__ qkv,
                                                       _Float16* __restrict__ ctxT) {
    __shared__ float red[8][1024];
    const int tid = threadIdx.x, lane = tid & 31, wave = tid >> 5;
    const int b = blockIdx.z, h = blockIdx.y;
    const _Float16* kb = qkv + ((size_t)b * OC_QKV + HID + h * DHEAD) * NPIX;
    const _Float16* vb = qkv + ((size_t)b * OC_QKV + 2 * HID + h * DHEAD) * NPIX;
    const int m = lane & 15, hf = lane >> 4;

    v8f acc[4] = {};                                  // [mi*2+ei]
    for (int it = 0; it < 16; ++it) {
        const int nb = wave * 512 + it * 32;
        if (it + 1 < 16) {                            // global_prefetch_b8
            __builtin_prefetch(kb + (size_t)m * NPIX + nb + 32, 0, 1);
            __builtin_prefetch(vb + (size_t)m * NPIX + nb + 32, 0, 1);
        }
        Frag fa[2], fb[2];
        #pragma unroll
        for (int mi = 0; mi < 2; ++mi) {
            const _Float16* s = kb + (size_t)(mi * 16 + m) * NPIX + nb + hf * 8;
            fa[mi].q[0] = *(const uint4*)s;
            fa[mi].q[1] = *(const uint4*)(s + 16);
        }
        #pragma unroll
        for (int ei = 0; ei < 2; ++ei) {
            const _Float16* s = vb + (size_t)(ei * 16 + m) * NPIX + nb + hf * 8;
            fb[ei].q[0] = *(const uint4*)s;
            fb[ei].q[1] = *(const uint4*)(s + 16);
        }
        #pragma unroll
        for (int mi = 0; mi < 2; ++mi)
            #pragma unroll
            for (int ei = 0; ei < 2; ++ei)
                acc[mi * 2 + ei] = WMMA_F16(fa[mi].v, fb[ei].v, acc[mi * 2 + ei]);
    }
    #pragma unroll
    for (int a = 0; a < 4; ++a)
        #pragma unroll
        for (int r = 0; r < 8; ++r)
            red[wave][a * 256 + r * 32 + lane] = acc[a][r];
    __syncthreads();
    for (int i = tid; i < 1024; i += 256) {           // cross-wave K reduction
        float s = 0.0f;
        #pragma unroll
        for (int w = 0; w < 8; ++w) s += red[w][i];
        const int a = i >> 8, rem = i & 255, r = rem >> 5, ln = rem & 31;
        const int d = (a >> 1) * 16 + r + ((ln >> 4) << 3);
        const int e = (a & 1) * 16 + (ln & 15);
        ctxT[((size_t)b * HEADS + h) * 1024 + e * DHEAD + d] = (_Float16)s;
    }
}

// ---------------------------------------------------------------------------
// attT[b, n, h*32+e] = sum_d ctxT[e,d] * qT[n,d].  M=e(32), K=32. No LDS.
__global__ __launch_bounds__(256) void att_gemm_kernel(const _Float16* __restrict__ qT,
                                                       const _Float16* __restrict__ ctxT,
                                                       _Float16* __restrict__ attT) {
    const int tid = threadIdx.x, lane = tid & 31, wave = tid >> 5;
    const int b = blockIdx.z, h = blockIdx.y, n0 = blockIdx.x * 256;
    const int m = lane & 15, hf = lane >> 4;
    const _Float16* cb = ctxT + ((size_t)b * HEADS + h) * 1024;
    const _Float16* qb = qT + ((size_t)b * HEADS + h) * (size_t)NPIX * DHEAD;

    Frag fa[2];
    #pragma unroll
    for (int mi = 0; mi < 2; ++mi) {                  // A = ctxT rows (K=d contig)
        const _Float16* s = cb + (mi * 16 + m) * DHEAD + hf * 8;
        fa[mi].q[0] = *(const uint4*)s;
        fa[mi].q[1] = *(const uint4*)(s + 16);
    }
    v8f acc[2][2] = {};
    #pragma unroll
    for (int nj = 0; nj < 2; ++nj) {                  // B = qT rows (K=d contig)
        const int n = n0 + wave * 32 + nj * 16 + m;
        Frag fb;
        fb.q[0] = *(const uint4*)(qb + (size_t)n * DHEAD + hf * 8);
        fb.q[1] = *(const uint4*)(qb + (size_t)n * DHEAD + 16 + hf * 8);
        #pragma unroll
        for (int mi = 0; mi < 2; ++mi)
            acc[mi][nj] = WMMA_F16(fa[mi].v, fb.v, acc[mi][nj]);
    }
    _Float16* ab = attT + (size_t)b * NPIX * HID;     // per-lane: fixed n, contig ch
    #pragma unroll
    for (int mi = 0; mi < 2; ++mi)
        #pragma unroll
        for (int nj = 0; nj < 2; ++nj) {
            const int n = n0 + wave * 32 + nj * 16 + m;
            union { _Float16 h[8]; uint4 u; } pk;
            #pragma unroll
            for (int r = 0; r < 8; ++r) pk.h[r] = (_Float16)acc[mi][nj][r];
            *(uint4*)(ab + (size_t)n * HID + h * DHEAD + mi * 16 + hf * 8) = pk.u;
        }
}

// ---------------------------------------------------------------------------
// out[b,o,n] = wo(256x128) @ attT + b_out; accumulate GN stats.  No GEMM LDS.
__global__ __launch_bounds__(256) void out_gemm_kernel(const _Float16* __restrict__ attT,
                                                       const _Float16* __restrict__ wo,
                                                       const float* __restrict__ bias,
                                                       float* __restrict__ out,
                                                       float* __restrict__ stats) {
    __shared__ float redS[256], redQ[256];
    const int tid = threadIdx.x, lane = tid & 31, wave = tid >> 5;
    const int wm = wave & 3, wn = wave >> 2;          // 4 waves along o, 2 along n
    const int b = blockIdx.z, o0 = blockIdx.y * 64, n0 = blockIdx.x * 128;
    const int m = lane & 15, hf = lane >> 4;
    const _Float16* ab = attT + (size_t)b * NPIX * HID;
    const _Float16* arow = wo + (size_t)(o0 + wm * 16 + m) * HID;

    v8f acc[4] = {};
    #pragma unroll
    for (int kk = 0; kk < HID; kk += 32) {
        Frag fa;                                      // A = wo rows (K=ch contig)
        fa.q[0] = *(const uint4*)(arow + kk + hf * 8);
        fa.q[1] = *(const uint4*)(arow + kk + 16 + hf * 8);
        #pragma unroll
        for (int nn = 0; nn < 4; ++nn) {              // B = attT rows (K=ch contig)
            const int n = n0 + wn * 64 + nn * 16 + m;
            Frag fb;
            fb.q[0] = *(const uint4*)(ab + (size_t)n * HID + kk + hf * 8);
            fb.q[1] = *(const uint4*)(ab + (size_t)n * HID + kk + 16 + hf * 8);
            acc[nn] = WMMA_F16(fa.v, fb.v, acc[nn]);
        }
    }
    float lsum = 0.0f, lsq = 0.0f;
    float* ob = out + ((size_t)b * C_IN + o0 + wm * 16) * NPIX;
    #pragma unroll
    for (int nn = 0; nn < 4; ++nn) {
        const int n = n0 + wn * 64 + nn * 16 + m;
        #pragma unroll
        for (int r = 0; r < 8; ++r) {
            const int mm = r + hf * 8;
            const float vv = acc[nn][r] + bias[o0 + wm * 16 + mm];
            ob[(size_t)mm * NPIX + n] = vv;
            lsum += vv; lsq += vv * vv;
        }
    }
    redS[tid] = lsum; redQ[tid] = lsq; __syncthreads();
    for (int s = 128; s > 0; s >>= 1) {
        if (tid < s) { redS[tid] += redS[tid + s]; redQ[tid] += redQ[tid + s]; }
        __syncthreads();
    }
    if (tid == 0) {
        atomicAdd(&stats[b * 2 + 0], redS[0]);
        atomicAdd(&stats[b * 2 + 1], redQ[0]);
    }
}

// ---------------------------------------------------------------------------
// GroupNorm(1 group) finalize, in place on d_out.
__global__ __launch_bounds__(256) void gn_finalize_kernel(float* __restrict__ out,
                                                          const float* __restrict__ stats,
                                                          const float* __restrict__ gamma,
                                                          const float* __restrict__ beta) {
    const size_t idx = (size_t)blockIdx.x * 256 + threadIdx.x;
    const size_t f = idx * 4;
    const int b = (int)(f >> 20);                     // C*N = 2^20
    const int c = (int)((f >> 12) & 255);             // N   = 2^12
    const float M = 1048576.0f;
    const float mean = stats[b * 2 + 0] / M;
    const float var  = stats[b * 2 + 1] / M - mean * mean;
    const float g = gamma[c] * rsqrtf(var + 1e-5f);
    const float be = beta[c];
    float4 v = *(float4*)(out + f);
    v.x = (v.x - mean) * g + be;
    v.y = (v.y - mean) * g + be;
    v.z = (v.z - mean) * g + be;
    v.w = (v.w - mean) * g + be;
    *(float4*)(out + f) = v;
}

// ---------------------------------------------------------------------------
extern "C" void kernel_launch(void* const* d_in, const int* in_sizes, int n_in,
                              void* d_out, int out_size, void* d_ws, size_t ws_size,
                              hipStream_t stream) {
    const float* x    = (const float*)d_in[0];
    const float* wqkv = (const float*)d_in[1];
    const float* wout = (const float*)d_in[2];
    const float* bout = (const float*)d_in[3];
    const float* gam  = (const float*)d_in[4];
    const float* bet  = (const float*)d_in[5];
    float* out = (float*)d_out;
    char* ws = (char*)d_ws;
    _Float16* xT   = (_Float16*)(ws + OFF_XT);
    _Float16* wqh  = (_Float16*)(ws + OFF_WQKV);
    _Float16* woh  = (_Float16*)(ws + OFF_WOUT);
    _Float16* qkv  = (_Float16*)(ws + OFF_QKV);
    _Float16* qT   = (_Float16*)(ws + OFF_QT);
    _Float16* ctxT = (_Float16*)(ws + OFF_CTX);
    _Float16* attT = (_Float16*)(ws + OFF_ATT);
    float*    stats = (float*)(ws + OFF_STATS);

    zero_stats_kernel<<<1, 64, 0, stream>>>(stats);
    pack_x_kernel<<<dim3(128, 8, 16), 256, 0, stream>>>(x, xT);
    pack_w_kernel<<<128, 256, 0, stream>>>(wqkv, wout, wqh, woh);
    qkv_gemm_kernel<<<dim3(64, 3, 16), 256, 0, stream>>>(xT, wqh, qkv);
    q_softmax_kernel<<<dim3(16, 4, 16), 256, 0, stream>>>(qkv, qT);
    k_softmax_kernel<<<2048, 256, 0, stream>>>(qkv);
    ctx_gemm_kernel<<<dim3(1, 4, 16), 256, 0, stream>>>(qkv, ctxT);
    att_gemm_kernel<<<dim3(16, 4, 16), 256, 0, stream>>>(qT, ctxT, attT);
    out_gemm_kernel<<<dim3(32, 4, 16), 256, 0, stream>>>(attT, woh, bout, out, stats);
    gn_finalize_kernel<<<16384, 256, 0, stream>>>(out, stats, gam, bet);
}
